// MetaPlasticProber_4346506903952
// MI455X (gfx1250) — matrix operand order repaired and checked by reading, not compiled, not run on hardware
//
#include <hip/hip_runtime.h>

// ---------------------------------------------------------------------------
// Types for CDNA5 WMMA (wave32): bf16 A/B fragments (16 bf16 = 8 VGPRs),
// f32 C/D accumulator (8 VGPRs).
// ---------------------------------------------------------------------------
typedef __attribute__((ext_vector_type(16))) __bf16 v16bf;
typedef __attribute__((ext_vector_type(8)))  float  v8f;

union Frag {
    uint4 u[2];
    v16bf v;
};

// 16-byte vector type matching the async-to-LDS builtin's parameter type
// (diagnostic: "int __vector(4) __device__ *").
typedef int b128_t __attribute__((vector_size(16)));

// ---------------------------------------------------------------------------
// CDNA5 async global->LDS copy (ASYNCcnt-tracked DMA, no VGPR staging).
// Guarded: falls back to a plain VGPR-staged copy if the builtin is absent
// (e.g. host pass, or a toolchain without gfx1250 async builtins).
// ---------------------------------------------------------------------------
#if __has_builtin(__builtin_amdgcn_global_load_async_to_lds_b128)
#define ASYNC_LDS 1
#else
#define ASYNC_LDS 0
#endif

__device__ __forceinline__ void async_cp16(const unsigned short* g, unsigned short* l) {
#if ASYNC_LDS
    __builtin_amdgcn_global_load_async_to_lds_b128(
        (__attribute__((address_space(1))) b128_t*)(g),
        (__attribute__((address_space(3))) b128_t*)(l),
        0, 0);
#else
    *(uint4*)l = *(const uint4*)g;
#endif
}

// wait until at most n async ops outstanding (n in {0,4,8})
__device__ __forceinline__ void wait_async(int n) {
#if __has_builtin(__builtin_amdgcn_s_wait_asynccnt)
    if (n == 0)      __builtin_amdgcn_s_wait_asynccnt(0);
    else if (n == 4) __builtin_amdgcn_s_wait_asynccnt(4);
    else             __builtin_amdgcn_s_wait_asynccnt(8);
#elif ASYNC_LDS
    if (n == 0)      asm volatile("s_wait_asynccnt 0x0" ::: "memory");
    else if (n == 4) asm volatile("s_wait_asynccnt 0x4" ::: "memory");
    else             asm volatile("s_wait_asynccnt 0x8" ::: "memory");
#else
    (void)n;   // sync fallback: nothing outstanding
#endif
}

// ---------------------------------------------------------------------------
// Helpers
// ---------------------------------------------------------------------------
__device__ __forceinline__ unsigned short f2bf(float f) {
    unsigned int u = __float_as_uint(f);
    unsigned int r = u + 0x7FFFu + ((u >> 16) & 1u);   // round-to-nearest-even
    return (unsigned short)(r >> 16);
}

__device__ __forceinline__ unsigned pcg(unsigned v) {
    v = v * 747796405u + 2891336453u;
    unsigned w = ((v >> ((v >> 28) + 4u)) ^ v) * 277803737u;
    return (w >> 22) ^ w;
}

// hash-based standard normal (stand-in for jax.random.normal; no runtime check)
__device__ __forceinline__ float gnoise(unsigned s) {
    unsigned a = pcg(s);
    unsigned b = pcg(s ^ 0x9e3779b9u);
    float u1 = ((a >> 8) + 1u) * (1.0f / 16777217.0f);
    float u2 = (b >> 8) * (1.0f / 16777216.0f);
    return sqrtf(-2.0f * logf(u1)) * cosf(6.28318530718f * u2);
}

__device__ __forceinline__ float sigm(float x) { return 1.0f / (1.0f + expf(-x)); }

// ---------------------------------------------------------------------------
// f32 -> bf16 conversion kernels
// ---------------------------------------------------------------------------
__global__ void k_f2bf(const float* __restrict__ in, unsigned short* __restrict__ out, int n) {
    int i = blockIdx.x * blockDim.x + threadIdx.x;
    if (i < n) out[i] = f2bf(in[i]);
}

// write [M,K] f32 into a wider bf16 matrix of leading-dim ldo at column coff
__global__ void k_f2bf_strided(const float* __restrict__ in, unsigned short* __restrict__ out,
                               int M, int K, int ldo, int coff) {
    int i = blockIdx.x * blockDim.x + threadIdx.x;
    if (i >= M * K) return;
    int m = i / K, k = i - m * K;
    out[(size_t)m * ldo + coff + k] = f2bf(in[i]);
}

// out[c*R + r] = bf(in[r*C + c])  (transpose + convert, for error @ pred_W)
__global__ void k_f2bf_transpose(const float* __restrict__ in, unsigned short* __restrict__ out,
                                 int R, int C) {
    int i = blockIdx.x * blockDim.x + threadIdx.x;
    if (i >= R * C) return;
    int r = i / C, c = i - r * C;
    out[(size_t)c * R + r] = f2bf(in[i]);
}

// ---------------------------------------------------------------------------
// bf16 WMMA GEMM:  C[M,N] = A[M,K] * B[N,K]^T (+bias), f32 accumulate.
// Block = 256 threads = 8 waves. Tile 128x128, K-step 32, 3-stage async
// global->LDS pipeline (2 tiles in flight under ASYNCcnt).
// Each wave: 64(M) x 32(N) => 4x2 accumulators of v8f, 8 WMMA per K-step.
// LDS rows padded to 40 bf16 (80B) so all fragment ds_load_b128 are 16B
// aligned and bank-spread. 3 stages x (A+B) = 60 KB LDS.
// Epilogue modes:
//   0: Cf = acc + bias
//   1: Cb = bf16(silu(acc + bias))                      (t = silu(hx@fW1^T+b1))
//   2: e = aux - (acc + bias); Cf = e; Cb = bf16(e)     (error = x - pred)
//   3: Cf = acc + bias; Cb = bf16(same) at col offset   (x, into hx right half)
// ---------------------------------------------------------------------------
#define NSTAGE 3

__global__ __launch_bounds__(256) void gemm_bf16_wmma(
    const unsigned short* __restrict__ A, int lda,
    const unsigned short* __restrict__ Bm, int ldb,
    const float* __restrict__ bias,
    float* __restrict__ Cf,
    unsigned short* __restrict__ Cb, int ldcb, int cboff,
    const float* __restrict__ aux,
    int M, int N, int K, int mode)
{
    __shared__ alignas(16) unsigned short As[NSTAGE][128 * 40];
    __shared__ alignas(16) unsigned short Bs[NSTAGE][128 * 40];

    const int tid  = threadIdx.x;
    const int lane = tid & 31;
    const int wv   = tid >> 5;        // wave 0..7
    const int wm   = wv >> 2;         // 0..1  (M)
    const int wn   = wv & 3;          // 0..3  (N)
    const int lh   = lane >> 4;       // lane half
    const int l15  = lane & 15;
    const int tileM = blockIdx.y * 128;
    const int tileN = blockIdx.x * 128;

    // per-thread tile-fill assignment: row r (0..127), 16-elem chunk c16 (0|16)
    const int r   = tid >> 1;
    const int c16 = (tid & 1) << 4;
    const unsigned short* gA = A  + (size_t)(tileM + r) * lda + c16;
    const unsigned short* gB = Bm + (size_t)(tileN + r) * ldb + c16;
    const int loff = r * 40 + c16;

    const int nk = K >> 5;            // K-steps of 32

    // prologue: fill up to NSTAGE stages (4 async b128 per thread per stage)
#pragma unroll
    for (int s = 0; s < NSTAGE; s++) {
        if (s < nk) {
            async_cp16(gA + s * 32,     &As[s][loff]);
            async_cp16(gA + s * 32 + 8, &As[s][loff + 8]);
            async_cp16(gB + s * 32,     &Bs[s][loff]);
            async_cp16(gB + s * 32 + 8, &Bs[s][loff + 8]);
        }
    }

    v8f acc[4][2];
#pragma unroll
    for (int i = 0; i < 4; i++)
#pragma unroll
        for (int j = 0; j < 2; j++)
#pragma unroll
            for (int t = 0; t < 8; t++) acc[i][j][t] = 0.0f;

    for (int kt = 0; kt < nk; kt++) {
        const int cur = kt % NSTAGE;
        int rem = nk - 1 - kt;                   // newer stages in flight
        if (rem > NSTAGE - 1) rem = NSTAGE - 1;
        wait_async(4 * rem);                     // oldest stage complete
        __syncthreads();

        Frag a[4], b[2];
#pragma unroll
        for (int i = 0; i < 4; i++) {
            int row = wm * 64 + i * 16 + l15;
            const char* p = (const char*)&As[cur][0] + row * 80 + lh * 16;
            a[i].u[0] = *(const uint4*)p;
            a[i].u[1] = *(const uint4*)(p + 32);
        }
#pragma unroll
        for (int j = 0; j < 2; j++) {
            int nn = wn * 32 + j * 16 + l15;
            const char* p = (const char*)&Bs[cur][0] + nn * 80 + lh * 16;
            b[j].u[0] = *(const uint4*)p;
            b[j].u[1] = *(const uint4*)(p + 32);
        }
#pragma unroll
        for (int i = 0; i < 4; i++)
#pragma unroll
            for (int j = 0; j < 2; j++)
                acc[i][j] = __builtin_amdgcn_wmma_f32_16x16x32_bf16(
                    false, a[i].v, false, b[j].v, (short)0, acc[i][j], false, false);

        __syncthreads();                         // all waves done reading stage cur

        const int knext = kt + NSTAGE;
        if (knext < nk) {                        // refill stage cur
            async_cp16(gA + knext * 32,     &As[cur][loff]);
            async_cp16(gA + knext * 32 + 8, &As[cur][loff + 8]);
            async_cp16(gB + knext * 32,     &Bs[cur][loff]);
            async_cp16(gB + knext * 32 + 8, &Bs[cur][loff + 8]);
        }
    }

    // epilogue: C/D layout -> lane l15 = N, VGPR v + 8*lh = M within 16x16 tile
#pragma unroll
    for (int i = 0; i < 4; i++) {
#pragma unroll
        for (int j = 0; j < 2; j++) {
            int n = tileN + wn * 32 + j * 16 + l15;
            float bv = bias ? bias[n] : 0.0f;
#pragma unroll
            for (int v = 0; v < 8; v++) {
                int m = tileM + wm * 64 + i * 16 + v + lh * 8;
                float val = acc[i][j][v] + bv;
                if (mode == 0) {
                    Cf[(size_t)m * N + n] = val;
                } else if (mode == 1) {
                    float s = val * sigm(val);
                    Cb[(size_t)m * ldcb + cboff + n] = f2bf(s);
                } else if (mode == 2) {
                    float e = aux[(size_t)m * N + n] - val;
                    Cf[(size_t)m * N + n] = e;
                    Cb[(size_t)m * ldcb + cboff + n] = f2bf(e);
                } else {
                    Cf[(size_t)m * N + n] = val;
                    Cb[(size_t)m * ldcb + cboff + n] = f2bf(val);
                }
            }
        }
    }
}

// ---------------------------------------------------------------------------
// Batch column mean: mean[j] = (1/M) * sum_b h[b][j]
// ---------------------------------------------------------------------------
__global__ void k_colmean(const float* __restrict__ h, float* __restrict__ mean, int M, int Hn) {
    int j = blockIdx.x * blockDim.x + threadIdx.x;
    if (j >= Hn) return;
    float s = 0.0f;
    for (int b = 0; b < M; b++) s += h[(size_t)b * Hn + j];
    mean[j] = s / (float)M;
}

// ---------------------------------------------------------------------------
// slow_comb / m_state precompute (single block): delta = h_mean (slow_base==0)
//   slow_vel  = 0.02*delta + 0.02*noise
//   slow_attr = 0.015*tanh(h_mean)
//   gain      = sigmoid(||delta||)
//   sc[j]     = (1.1*slow_vel + 0.35*slow_attr) * gain
//   mt[j]     = 0.25 * sigmoid(beta_m) * h_mean[j]
// ---------------------------------------------------------------------------
__global__ void k_prep_slow(const float* __restrict__ hmean, const float* __restrict__ bm_a,
                            int layer, float* __restrict__ sc, float* __restrict__ mt, int Hn) {
    __shared__ float red[256];
    int tid = threadIdx.x;
    float ss = 0.0f;
    for (int j = tid; j < Hn; j += 256) { float d = hmean[j]; ss += d * d; }
    red[tid] = ss;
    __syncthreads();
    for (int s = 128; s > 0; s >>= 1) {
        if (tid < s) red[tid] += red[tid + s];
        __syncthreads();
    }
    float gain = sigm(sqrtf(red[0]));
    float sbm  = sigm(bm_a[layer]);
    for (int j = tid; j < Hn; j += 256) {
        float d = hmean[j];
        float noise = gnoise((unsigned)j ^ (0x85ebca6bu * (unsigned)(layer + 1)) ^ 0xc2b2ae35u);
        float slow_vel  = 0.02f * d + 0.02f * noise;
        float slow_attr = 0.015f * tanhf(d);
        sc[j] = (1.1f * slow_vel + 0.35f * slow_attr) * gain;
        mt[j] = 0.25f * sbm * d;
    }
}

// ---------------------------------------------------------------------------
// MetaPlastic cell elementwise + row reductions. One block per batch row.
//   S = silu(h); dS = (1-eta)/dt * S
//   meta_trace = 0.3*(dS - sc)
//   proj = <meta_trace, dS> * dS          (row reduction 1)
//   meta_dir  = meta_trace - 0.6*proj + 0.2*noise
//   pre = decay*h + (1-decay)*fx + alpha*ep + mt + 1.8*sbm*beta*meta_dir
//   hn  = 0.72*h + 0.28*tanh(pre)
//   out = hn * sigmoid(2 - ||hn||)        (row reduction 2)
// ---------------------------------------------------------------------------
__global__ __launch_bounds__(256) void k_cell(
    const float* __restrict__ h,  const float* __restrict__ fx,
    const float* __restrict__ ep, const float* __restrict__ sc,
    const float* __restrict__ mt,
    const float* __restrict__ alpha_a, const float* __restrict__ beta_a,
    const float* __restrict__ lam_a,   const float* __restrict__ bm_a, int layer,
    float* __restrict__ outf, unsigned short* __restrict__ outb,
    int ldb2, int cboff, int Hn)
{
    __shared__ float red[256];
    const int b   = blockIdx.x;
    const int tid = threadIdx.x;
    const float alpha = alpha_a[layer];
    const float beta  = beta_a[layer];
    const float decay = expf(-fabsf(lam_a[layer]) * 0.06f);
    const float sbm   = sigm(bm_a[layer]);
    const float dsc   = (1.0f - 0.86f) / 0.06f;

    float hv[4], dS[4], mtr[4], fxv[4], epv[4], mts[4];
    float dot = 0.0f;
#pragma unroll
    for (int k = 0; k < 4; k++) {
        int j = tid + k * 256;
        size_t idx = (size_t)b * Hn + j;
        float x = h[idx];  hv[k] = x;
        float S = x * sigm(x);
        float d = dsc * S; dS[k] = d;
        float m = 0.3f * (d - sc[j]); mtr[k] = m;
        dot += m * d;
        fxv[k] = fx[idx];
        epv[k] = ep[idx];
        mts[k] = mt[j];
    }
    red[tid] = dot;
    __syncthreads();
    for (int s = 128; s > 0; s >>= 1) {
        if (tid < s) red[tid] += red[tid + s];
        __syncthreads();
    }
    float rowdot = red[0];
    __syncthreads();

    float hn[4], nsum = 0.0f;
#pragma unroll
    for (int k = 0; k < 4; k++) {
        int j = tid + k * 256;
        float noise = gnoise(((unsigned)b * 1024u + (unsigned)j) ^ (0x9e3779b9u * (unsigned)(layer + 1)));
        float mdir  = mtr[k] - 0.6f * rowdot * dS[k] + 0.2f * noise;
        float pre   = decay * hv[k] + (1.0f - decay) * fxv[k] + alpha * epv[k]
                    + mts[k] + 1.8f * sbm * beta * mdir;
        float v = 0.72f * hv[k] + 0.28f * tanhf(pre);
        hn[k] = v;
        nsum += v * v;
    }
    red[tid] = nsum;
    __syncthreads();
    for (int s = 128; s > 0; s >>= 1) {
        if (tid < s) red[tid] += red[tid + s];
        __syncthreads();
    }
    float norm  = sqrtf(red[0]) + 1e-6f;
    float scale = sigm(2.0f - norm);
#pragma unroll
    for (int k = 0; k < 4; k++) {
        int j = tid + k * 256;
        size_t idx = (size_t)b * Hn + j;
        float o = hn[k] * scale;
        outf[idx] = o;
        outb[(size_t)b * ldb2 + cboff + j] = f2bf(o);
    }
}

// ---------------------------------------------------------------------------
// Orchestration
// ---------------------------------------------------------------------------
extern "C" void kernel_launch(void* const* d_in, const int* in_sizes, int n_in,
                              void* d_out, int out_size, void* d_ws, size_t ws_size,
                              hipStream_t stream) {
    (void)in_sizes; (void)n_in; (void)out_size; (void)ws_size;
    const int Bm = 4096, Hh = 1024, Gg = 768;

    const float* base  = (const float*)d_in[0];
    const float* h0    = (const float*)d_in[1];
    const float* projW = (const float*)d_in[2];
    const float* projb = (const float*)d_in[3];
    const float* fW1   = (const float*)d_in[4];
    const float* fb1   = (const float*)d_in[5];
    const float* fW2   = (const float*)d_in[6];
    const float* fb2   = (const float*)d_in[7];
    const float* pW    = (const float*)d_in[8];
    const float* pb    = (const float*)d_in[9];
    // d_in[10], d_in[11] (gate_W, gate_b): unused -- memory_context is exactly 0
    const float* alpha = (const float*)d_in[12];
    const float* beta  = (const float*)d_in[13];
    const float* lam   = (const float*)d_in[14];
    const float* bm    = (const float*)d_in[15];

    // ---- workspace layout (all offsets 256B aligned) ----
    char* w = (char*)d_ws;
    size_t off = 0;
    auto alloc = [&](size_t bytes) { void* p = w + off; off += (bytes + 255) & ~(size_t)255; return p; };

    float* x_f   = (float*)alloc((size_t)Bm * Hh * 4);
    float* err_f = (float*)alloc((size_t)Bm * Hh * 4);
    float* fx_f  = (float*)alloc((size_t)Bm * Hh * 4);
    float* ep_f  = (float*)alloc((size_t)Bm * Hh * 4);
    unsigned short* hx_b   = (unsigned short*)alloc((size_t)Bm * 2 * Hh * 2);
    unsigned short* t_b    = (unsigned short*)alloc((size_t)Bm * 2 * Hh * 2);
    unsigned short* err_b  = (unsigned short*)alloc((size_t)Bm * Hh * 2);
    unsigned short* base_b = (unsigned short*)alloc((size_t)Bm * Gg * 2);
    unsigned short* projW_b = (unsigned short*)alloc((size_t)Hh * Gg * 2);
    unsigned short* fW1_b   = (unsigned short*)alloc((size_t)2 * Hh * 2 * Hh * 2);
    unsigned short* fW2_b   = (unsigned short*)alloc((size_t)Hh * 2 * Hh * 2);
    unsigned short* pW_b    = (unsigned short*)alloc((size_t)Hh * Hh * 2);
    unsigned short* pWT_b   = (unsigned short*)alloc((size_t)Hh * Hh * 2);
    float* hmean = (float*)alloc(Hh * 4);
    float* sc    = (float*)alloc(Hh * 4);
    float* mt    = (float*)alloc(Hh * 4);

    dim3 blk(256);
    auto cvt = [&](const float* in, unsigned short* out, int n) {
        k_f2bf<<<(n + 255) / 256, blk, 0, stream>>>(in, out, n);
    };
    auto gemm = [&](const unsigned short* A, int lda, const unsigned short* Bmat, int ldb,
                    const float* bias, float* Cf, unsigned short* Cb, int ldcb, int cboff,
                    const float* aux, int M, int N, int K, int mode) {
        dim3 grid(N / 128, M / 128);
        gemm_bf16_wmma<<<grid, blk, 0, stream>>>(A, lda, Bmat, ldb, bias,
                                                 Cf, Cb, ldcb, cboff, aux, M, N, K, mode);
    };

    // x = base @ projW^T + projb  -> x_f (f32) and hx[:, H:2H] (bf16)
    cvt(base, base_b, Bm * Gg);
    cvt(projW, projW_b, Hh * Gg);
    gemm(base_b, Gg, projW_b, Gg, projb, x_f, hx_b, 2 * Hh, Hh, nullptr,
         Bm, Hh, Gg, /*mode=*/3);

    for (int i = 0; i < 3; i++) {
        const float* h_i = h0 + (size_t)i * Bm * Hh;

        // h0[i] -> hx[:, 0:H] (bf16); per-layer weights -> bf16
        k_f2bf_strided<<<(Bm * Hh + 255) / 256, blk, 0, stream>>>(h_i, hx_b, Bm, Hh, 2 * Hh, 0);
        cvt(fW1 + (size_t)i * 2 * Hh * 2 * Hh, fW1_b, 2 * Hh * 2 * Hh);
        cvt(fW2 + (size_t)i * Hh * 2 * Hh,     fW2_b, Hh * 2 * Hh);
        cvt(pW  + (size_t)i * Hh * Hh,         pW_b,  Hh * Hh);
        k_f2bf_transpose<<<(Hh * Hh + 255) / 256, blk, 0, stream>>>(
            pW + (size_t)i * Hh * Hh, pWT_b, Hh, Hh);

        // error = x - (h @ pW^T + pb)           [mode 2: f32 + bf16]
        gemm(hx_b, 2 * Hh, pW_b, Hh, pb + i * Hh, err_f, err_b, Hh, 0, x_f,
             Bm, Hh, Hh, 2);
        // t = silu(hx @ fW1^T + fb1)            [mode 1: bf16]
        gemm(hx_b, 2 * Hh, fW1_b, 2 * Hh, fb1 + i * 2 * Hh, nullptr, t_b, 2 * Hh, 0,
             nullptr, Bm, 2 * Hh, 2 * Hh, 1);
        // fx = t @ fW2^T + fb2                  [mode 0: f32]
        gemm(t_b, 2 * Hh, fW2_b, 2 * Hh, fb2 + i * Hh, fx_f, nullptr, 0, 0,
             nullptr, Bm, Hh, 2 * Hh, 0);
        // ep = error @ pW  (B = pW^T in [N][K]) [mode 0: f32]
        gemm(err_b, Hh, pWT_b, Hh, nullptr, ep_f, nullptr, 0, 0,
             nullptr, Bm, Hh, Hh, 0);

        // batch reductions + slow-state precompute
        k_colmean<<<(Hh + 255) / 256, blk, 0, stream>>>(h_i, hmean, Bm, Hh);
        k_prep_slow<<<1, blk, 0, stream>>>(hmean, bm, i, sc, mt, Hh);

        // cell update -> f32 x (or final out) + bf16 into hx[:, H:2H]
        float* outf = (i == 2) ? (float*)d_out : x_f;
        k_cell<<<Bm, blk, 0, stream>>>(h_i, fx_f, ep_f, sc, mt,
                                       alpha, beta, lam, bm, i,
                                       outf, hx_b, 2 * Hh, Hh, Hh);
    }
}
